// MaskablePPOPolicy_31233002176979
// MI455X (gfx1250) — matrix operand order repaired and checked by reading, not compiled, not run on hardware
//
#include <hip/hip_runtime.h>

// ---------------- problem constants ----------------
#define EMB   256
#define MAXN  512
#define NB    128
#define NTOT  (NB * MAXN)          // 65536 rows
#define KCAT  512                  // [x | h] concatenated K
#define G4    1024                 // 4*H gate dims
#define FEATW 259                  // features row width

// LDS row strides (halves), padded so lane l16 starts at 4*l16 dwords mod 64
#define LDS2_STRIDE 520            // kernel2: 128 rows -> 133120 B
#define LDS4_STRIDE 264            // kernel4: 256 rows -> 135168 B

typedef _Float16 v16h __attribute__((ext_vector_type(16)));
typedef _Float16 v8h  __attribute__((ext_vector_type(8)));
typedef float    v8f  __attribute__((ext_vector_type(8)));

union AU { v16h v; v8h h[2]; };

// ---------------- workspace layout (bytes) ----------------
static constexpr size_t OFF_XH    = 0;                                    // N x 512 f16
static constexpr size_t OFF_WP    = OFF_XH   + (size_t)NTOT * KCAT * 2;   // 1024 x 512 f16
static constexpr size_t OFF_W7H   = OFF_WP   + (size_t)G4 * KCAT * 2;     // 256 x 256 f16
static constexpr size_t OFF_BIASG = OFF_W7H  + (size_t)EMB * EMB * 2;     // 1024 f32
static constexpr size_t OFF_HF16  = OFF_BIASG + (size_t)G4 * 4;           // N x 256 f16
static constexpr size_t OFF_POOL  = OFF_HF16 + (size_t)NTOT * EMB * 2;    // 128 x 256 f16
static constexpr size_t OFF_GTERM = OFF_POOL + (size_t)NB * EMB * 2;      // 128 f32

__device__ __forceinline__ float sigf(float x) {
    return 1.0f / (1.0f + __expf(-x));
}

// ---------------- prep: pack [x|h] activations to f16 ----------------
__global__ void __launch_bounds__(256) prep_xh_kernel(
    const float* __restrict__ features, const float* __restrict__ h0,
    const int* __restrict__ terminal, _Float16* __restrict__ Xh)
{
    size_t idx = (size_t)blockIdx.x * blockDim.x + threadIdx.x;
    if (idx >= (size_t)NTOT * KCAT) return;
    int n = (int)(idx >> 9);
    int k = (int)(idx & 511);
    float v;
    if (k < EMB) {
        v = features[(size_t)n * FEATW + k];               // mu_raw
    } else {
        v = terminal[n >> 9] ? 0.0f : h0[(size_t)n * EMB + (k - EMB)];
    }
    Xh[idx] = (_Float16)v;
}

// ---------------- prep: pack weights to f16, sum biases ----------------
__global__ void __launch_bounds__(256) prep_w_kernel(
    const float* __restrict__ w_ih, const float* __restrict__ w_hh,
    const float* __restrict__ b_ih, const float* __restrict__ b_hh,
    const float* __restrict__ w7,
    _Float16* __restrict__ Wp, float* __restrict__ biasg,
    _Float16* __restrict__ W7h)
{
    int idx = blockIdx.x * blockDim.x + threadIdx.x;
    if (idx < G4 * KCAT) {
        int j = idx >> 9, k = idx & 511;
        float v = (k < EMB) ? w_ih[(size_t)j * EMB + k]
                            : w_hh[(size_t)j * EMB + (k - EMB)];
        Wp[idx] = (_Float16)v;
    }
    if (idx < G4) biasg[idx] = b_ih[idx] + b_hh[idx];
    if (idx < EMB * EMB) W7h[idx] = (_Float16)w7[idx];
}

// ---------------- main: gates GEMM + fused LSTM cell ----------------
// block: 8 waves = 8 x 16-row tiles (128 rows), ONE 32-h-col group.
// The block's 128 gate columns x K=512 of weights (128 KiB f16) are staged
// to LDS once; each wave hoists its full A row (K=512) into registers, so
// the main loop is a pure ds_load + v_wmma stream.
__global__ void __launch_bounds__(256) gates_lstm_kernel(
    const _Float16* __restrict__ Xh, const _Float16* __restrict__ Wp,
    const float* __restrict__ biasg, const float* __restrict__ c0,
    const int* __restrict__ terminal, _Float16* __restrict__ Hf16)
{
    extern __shared__ _Float16 wl[];    // 128 x LDS2_STRIDE halves

    const int wave = threadIdx.x >> 5;
    const int lane = threadIdx.x & 31;
    const int half = lane >> 4;         // 0: lanes 0-15, 1: lanes 16-31
    const int l16  = lane & 15;
    const int rb   = blockIdx.x >> 3;   // row block (128 rows)
    const int cg   = blockIdx.x & 7;    // column group: 32 h-cols
    const int m0   = (rb * 8 + wave) * 16;

    // ---- hoist full A row into registers (16 x 2 b128 loads, one clause) ----
    const _Float16* arow = Xh + (size_t)(m0 + l16) * KCAT;
    AU a[16];
#pragma unroll
    for (int i = 0; i < 16; ++i) {
        a[i].h[0] = *(const v8h*)(arow + i * 32 + half * 8);
        a[i].h[1] = *(const v8h*)(arow + i * 32 + half * 8 + 16);
    }

    // ---- stage the block's weight slab to LDS (128 rows x 512 halves) ----
    // LDS row = t*16 + col, where t = gate*2 + sub16, col = 0..15
#pragma unroll
    for (int it = 0; it < 32; ++it) {
        int c   = it * 256 + threadIdx.x;      // 8192 chunks of 8 halves
        int row = c >> 6;                      // 0..127
        int k   = (c & 63) * 8;                // 0..504
        int t   = row >> 4;
        int col = row & 15;
        int gcol = (t >> 1) * EMB + cg * 32 + (t & 1) * 16 + col;
        *(v8h*)&wl[row * LDS2_STRIDE + k] =
            *(const v8h*)(Wp + (size_t)gcol * KCAT + k);
    }
    __syncthreads();

    // ---- accumulators preloaded with bias (per gate-column == per lane) ----
    v8f acc[8];
#pragma unroll
    for (int t = 0; t < 8; ++t) {
        int gcol = (t >> 1) * EMB + cg * 32 + (t & 1) * 16 + l16;
        float bv = biasg[gcol];
        acc[t] = (v8f){bv, bv, bv, bv, bv, bv, bv, bv};
    }

    // ---- main loop: 16 K-steps x 8 tiles = 128 WMMAs, B from LDS ----
#pragma unroll
    for (int i = 0; i < 16; ++i) {
        const int kb = i * 32;
#pragma unroll
        for (int t = 0; t < 8; ++t) {
            const _Float16* bp = &wl[(t * 16 + l16) * LDS2_STRIDE + kb + half * 8];
            AU b;
            b.h[0] = *(const v8h*)(bp);
            b.h[1] = *(const v8h*)(bp + 16);
            acc[t] = __builtin_amdgcn_wmma_f32_16x16x32_f16(
                false, a[i].v, false, b.v, (short)0, acc[t], false, false);
        }
    }

    // ---- fused LSTM cell epilogue (tile rows share one batch) ----
    const int termF = terminal[m0 >> 9];
#pragma unroll
    for (int s = 0; s < 2; ++s) {
        const int jc = cg * 32 + s * 16 + l16;     // h column
#pragma unroll
        for (int r = 0; r < 8; ++r) {
            const int n = m0 + half * 8 + r;       // C layout: M = r + 8*half
            float c0v = termF ? 0.0f : c0[(size_t)n * EMB + jc];
            float iv = sigf(acc[0 * 2 + s][r]);
            float fv = sigf(acc[1 * 2 + s][r]);
            float gv = tanhf(acc[2 * 2 + s][r]);
            float ov = sigf(acc[3 * 2 + s][r]);
            float cn = fv * c0v + iv * gv;
            float hn = ov * tanhf(cn);
            Hf16[(size_t)n * EMB + jc] = (_Float16)hn;
        }
    }
}

// ---------------- mean pool: contiguous 512-row blocks ----------------
__global__ void __launch_bounds__(256) pool_kernel(
    const _Float16* __restrict__ Hf16, _Float16* __restrict__ pool)
{
    int b = blockIdx.x, c = threadIdx.x;
    const _Float16* p = Hf16 + (size_t)b * MAXN * EMB + c;
    float s = 0.0f;
    for (int r = 0; r < MAXN; ++r) s += (float)p[(size_t)r * EMB];
    pool[b * EMB + c] = (_Float16)(s * (1.0f / (float)MAXN));
}

// ---------------- per-batch scalar: relu(pool@w6^T+b6) . w5[:256] --------
__global__ void __launch_bounds__(256) gterm_kernel(
    const _Float16* __restrict__ pool, const float* __restrict__ w6,
    const float* __restrict__ b6, const float* __restrict__ w5,
    float* __restrict__ gterm)
{
    __shared__ float red[256];
    int b = blockIdx.x, j = threadIdx.x;
    const _Float16* pb = pool + b * EMB;
    const float* wj = w6 + (size_t)j * EMB;
    float acc = b6[j];
    for (int k = 0; k < EMB; ++k) acc += (float)pb[k] * wj[k];
    red[j] = fmaxf(acc, 0.0f) * w5[j];
    __syncthreads();
    for (int o = 128; o > 0; o >>= 1) {
        if (j < o) red[j] += red[j + o];
        __syncthreads();
    }
    if (j == 0) gterm[b] = red[0];
}

// ---------------- local_action GEMM + fused relu.w5 + mask ----------------
// block: 8 waves = 128 rows; all of w7 (256x256 f16) staged to LDS; each
// wave hoists its full A row (K=256) and sweeps 16 column tiles.
__global__ void __launch_bounds__(256) act_logits_kernel(
    const _Float16* __restrict__ Hf16, const _Float16* __restrict__ W7h,
    const float* __restrict__ b7, const float* __restrict__ w5,
    const float* __restrict__ b5, const float* __restrict__ gterm,
    const float* __restrict__ features, float* __restrict__ out)
{
    extern __shared__ _Float16 wl4[];   // 256 x LDS4_STRIDE halves

    const int wave = threadIdx.x >> 5;
    const int lane = threadIdx.x & 31;
    const int half = lane >> 4;
    const int l16  = lane & 15;
    const int m0   = (blockIdx.x * 8 + wave) * 16;

    // ---- hoist full A row (K=256) ----
    const _Float16* arow = Hf16 + (size_t)(m0 + l16) * EMB;
    AU a[8];
#pragma unroll
    for (int i = 0; i < 8; ++i) {
        a[i].h[0] = *(const v8h*)(arow + i * 32 + half * 8);
        a[i].h[1] = *(const v8h*)(arow + i * 32 + half * 8 + 16);
    }

    // ---- stage all of w7 (256 rows x 256 halves) to LDS ----
#pragma unroll
    for (int it = 0; it < 32; ++it) {
        int c   = it * 256 + threadIdx.x;      // 8192 chunks of 8 halves
        int row = c >> 5;                      // 0..255 (= output column)
        int k   = (c & 31) * 8;                // 0..248
        *(v8h*)&wl4[row * LDS4_STRIDE + k] =
            *(const v8h*)(W7h + (size_t)row * EMB + k);
    }
    __syncthreads();

    float rowsum[8] = {0, 0, 0, 0, 0, 0, 0, 0};

#pragma unroll 1
    for (int ct = 0; ct < 16; ++ct) {
        const int col = ct * 16 + l16;
        float bv = b7[col];
        v8f acc = (v8f){bv, bv, bv, bv, bv, bv, bv, bv};
#pragma unroll
        for (int i = 0; i < 8; ++i) {
            const _Float16* bp = &wl4[col * LDS4_STRIDE + i * 32 + half * 8];
            AU b;
            b.h[0] = *(const v8h*)(bp);
            b.h[1] = *(const v8h*)(bp + 16);
            acc = __builtin_amdgcn_wmma_f32_16x16x32_f16(
                false, a[i].v, false, b.v, (short)0, acc, false, false);
        }
        float wv = w5[EMB + col];              // w5 high half
#pragma unroll
        for (int r = 0; r < 8; ++r)
            rowsum[r] += fmaxf(acc[r], 0.0f) * wv;
    }

    // reduce across the 16 lanes holding the 16 columns of each row half
#pragma unroll
    for (int r = 0; r < 8; ++r) {
#pragma unroll
        for (int o = 1; o < 16; o <<= 1)
            rowsum[r] += __shfl_xor(rowsum[r], o, 32);
    }

    if (l16 == 0) {
        float gt = gterm[m0 >> 9] + b5[0];
#pragma unroll
        for (int r = 0; r < 8; ++r) {
            int n = m0 + half * 8 + r;
            bool reach = features[(size_t)n * FEATW + (EMB + 1)] > 0.5f;
            out[n] = reach ? (rowsum[r] + gt) : -__builtin_huge_valf();
        }
    }
}

extern "C" void kernel_launch(void* const* d_in, const int* in_sizes, int n_in,
                              void* d_out, int out_size, void* d_ws, size_t ws_size,
                              hipStream_t stream) {
    const float* features = (const float*)d_in[0];
    const float* h0       = (const float*)d_in[1];
    const float* c0       = (const float*)d_in[2];
    const float* w_ih     = (const float*)d_in[3];
    const float* w_hh     = (const float*)d_in[4];
    const float* b_ih     = (const float*)d_in[5];
    const float* b_hh     = (const float*)d_in[6];
    const float* w5       = (const float*)d_in[7];
    const float* b5       = (const float*)d_in[8];
    const float* w6       = (const float*)d_in[9];
    const float* b6       = (const float*)d_in[10];
    const float* w7       = (const float*)d_in[11];
    const float* b7       = (const float*)d_in[12];
    const int*   terminal = (const int*)d_in[13];
    // d_in[14] batch_data (int64) unused: batches are contiguous 512-row blocks
    float* out = (float*)d_out;

    char* ws = (char*)d_ws;
    _Float16* Xh    = (_Float16*)(ws + OFF_XH);
    _Float16* Wp    = (_Float16*)(ws + OFF_WP);
    _Float16* W7h   = (_Float16*)(ws + OFF_W7H);
    float*    biasg = (float*)(ws + OFF_BIASG);
    _Float16* Hf16  = (_Float16*)(ws + OFF_HF16);
    _Float16* pool  = (_Float16*)(ws + OFF_POOL);
    float*    gterm = (float*)(ws + OFF_GTERM);

    // 1) pack activations and weights to f16
    {
        size_t tot = (size_t)NTOT * KCAT;
        int blocks = (int)((tot + 255) / 256);
        prep_xh_kernel<<<blocks, 256, 0, stream>>>(features, h0, terminal, Xh);
    }
    {
        int blocks = (G4 * KCAT + 255) / 256;
        prep_w_kernel<<<blocks, 256, 0, stream>>>(w_ih, w_hh, b_ih, b_hh, w7,
                                                  Wp, biasg, W7h);
    }
    // 2) fused gates GEMM (K=512) + LSTM cell -> mu_mem (f16)
    {
        int blocks = (NTOT / 128) * 8;                     // 4096
        size_t lds = (size_t)128 * LDS2_STRIDE * 2;        // 133120 B
        gates_lstm_kernel<<<blocks, 256, lds, stream>>>(Xh, Wp, biasg, c0,
                                                        terminal, Hf16);
    }
    // 3) segment mean pool (contiguous blocks)
    pool_kernel<<<NB, 256, 0, stream>>>(Hf16, pool);
    // 4) per-batch global-state scalar
    gterm_kernel<<<NB, 256, 0, stream>>>(pool, w6, b6, w5, gterm);
    // 5) local_action GEMM + relu + w5 dot + mask -> logits
    {
        int blocks = NTOT / 128;                           // 512
        size_t lds = (size_t)EMB * LDS4_STRIDE * 2;        // 135168 B
        act_logits_kernel<<<blocks, 256, lds, stream>>>(
            Hf16, W7h, b7, w5, b5, gterm, features, out);
    }
}
